// InferenceModel_77773267796379
// MI455X (gfx1250) — matrix operand order repaired and checked by reading, not compiled, not run on hardware
//
#include <hip/hip_runtime.h>
#include <cmath>

typedef __attribute__((ext_vector_type(16))) _Float16 v16h;
typedef __attribute__((ext_vector_type(8)))  _Float16 v8h;
typedef __attribute__((ext_vector_type(8)))  float    v8f;

__device__ __forceinline__ float gelu_exact(float x) {
    return 0.5f * x * (1.0f + erff(x * 0.70710678118654752f));
}

// ---- transpose + f16-convert the four 512x512 hidden weights: Wt[l][n][k] = W_l[k][n]
__global__ void k_transpose_w(const float* __restrict__ W1, const float* __restrict__ W2,
                              const float* __restrict__ W3, const float* __restrict__ W4,
                              _Float16* __restrict__ Wt) {
    int idx = blockIdx.x * blockDim.x + threadIdx.x;    // 4*512*512 threads
    int l   = idx >> 18;
    int rem = idx & 262143;
    int k   = rem >> 9;
    int n   = rem & 511;
    const float* W = (l == 0) ? W1 : (l == 1) ? W2 : (l == 2) ? W3 : W4;
    Wt[(size_t)l * 262144 + (size_t)n * 512 + k] = (_Float16)W[k * 512 + n];
}

// ---- constant part of first layer: t_emb @ W_in[14:526] + b_in   (t_emb same for all rows)
__global__ void k_const_in(const float* __restrict__ W_in, const float* __restrict__ b_in,
                           float* __restrict__ const_in, float cur_t) {
    __shared__ float emb[512];
    int j = threadIdx.x;                                 // 512 threads, 1 block
    int i = j >> 1;
    float di  = expf((float)(2 * i) * (-9.210340371976184f / 512.0f)); // -log(1e4)/512
    float ang = cur_t * di;
    emb[j] = (j & 1) ? cosf(ang) : sinf(ang);
    __syncthreads();
    float s = b_in[j];
    for (int r = 0; r < 512; ++r)
        s += emb[r] * W_in[(14 + r) * 512 + j];
    const_in[j] = s;
}

// ---- first layer: K collapsed to 14 (state 6 + action 8) + constant vector; gelu; f16 out
__global__ void k_first_layer(const float* __restrict__ x_pre, const float* __restrict__ noise_s,
                              const float* __restrict__ W_in, const float* __restrict__ const_in,
                              _Float16* __restrict__ H) {
    int idx = blockIdx.x * blockDim.x + threadIdx.x;     // 2048*512 threads
    int j = idx & 511;
    int b = idx >> 9;
    float s = const_in[j];
#pragma unroll
    for (int i = 0; i < 6; ++i) s += x_pre[b * 6 + i] * W_in[i * 512 + j];
#pragma unroll
    for (int i = 0; i < 8; ++i) s += noise_s[b * 8 + i] * W_in[(6 + i) * 512 + j];
    H[(size_t)b * 512 + j] = (_Float16)gelu_exact(s);
}

// ---- WMMA GEMM: Y = gelu(X(2048x512,f16) @ W(512x512) + b), W pre-transposed (N-major)
// 128 threads = 4 waves; each wave owns a 32x64 output tile (2 M-tiles x 4 N-tiles)
// so every B fragment feeds two WMMAs: 12 b128 loads per 8 WMMAs.
__global__ __launch_bounds__(128)
void k_gemm512(const _Float16* __restrict__ X, const _Float16* __restrict__ Wt,
               const float* __restrict__ bias, _Float16* __restrict__ Y) {
    const int lane = threadIdx.x & 31;
    const int wave = threadIdx.x >> 5;
    const int g    = lane >> 4;       // lane group (K interleave select)
    const int ln   = lane & 15;
    const int rowM = blockIdx.x * 128 + wave * 32;       // two 16-row tiles: rowM, rowM+16
    const int colB = blockIdx.y * 64;

    v8f acc[2][4];
#pragma unroll
    for (int ct = 0; ct < 4; ++ct) {
        float bv = bias[colB + ct * 16 + ln];            // bias depends only on column
#pragma unroll
        for (int r = 0; r < 8; ++r) { acc[0][ct][r] = bv; acc[1][ct][r] = bv; }
    }

    const _Float16* xrow0 = X + (size_t)(rowM + ln) * 512;
    const _Float16* xrow1 = X + (size_t)(rowM + 16 + ln) * 512;
#pragma unroll 2
    for (int kc = 0; kc < 512; kc += 32) {
        // A fragments (16x32 f16): lanes 0-15 rows, e0..15 = K kc+g*16.. (contiguous)
        v8h lo0 = *(const v8h*)(xrow0 + kc + g * 16);
        v8h hi0 = *(const v8h*)(xrow0 + kc + g * 16 + 8);
        v8h lo1 = *(const v8h*)(xrow1 + kc + g * 16);
        v8h hi1 = *(const v8h*)(xrow1 + kc + g * 16 + 8);
        v16h a0, a1;
#pragma unroll
        for (int i = 0; i < 8; ++i) {
            a0[i] = lo0[i]; a0[8 + i] = hi0[i];
            a1[i] = lo1[i]; a1[8 + i] = hi1[i];
        }
#pragma unroll
        for (int ct = 0; ct < 4; ++ct) {
            // B fragment (32x16 f16): lane -> column n, 16 contiguous K at g*16
            const _Float16* wrow = Wt + (size_t)(colB + ct * 16 + ln) * 512 + kc + g * 16;
            v16h b = *(const v16h*)wrow;
            acc[0][ct] = __builtin_amdgcn_wmma_f32_16x16x32_f16(
                false, a0, false, b, (short)0, acc[0][ct], false, false);
            acc[1][ct] = __builtin_amdgcn_wmma_f32_16x16x32_f16(
                false, a1, false, b, (short)0, acc[1][ct], false, false);
        }
    }

#pragma unroll
    for (int mt = 0; mt < 2; ++mt) {
#pragma unroll
        for (int ct = 0; ct < 4; ++ct) {
            int col = colB + ct * 16 + ln;
#pragma unroll
            for (int r = 0; r < 8; ++r) {
                int mm = g * 8 + r;                      // C/D layout: VGPR r, lane group -> row
                Y[(size_t)(rowM + mt * 16 + mm) * 512 + col] =
                    (_Float16)gelu_exact(acc[mt][ct][r]);
            }
        }
    }
}

// ---- output head: out8 = H(f16) @ W_out(512x8,f32) + b_out
__global__ void k_out_proj(const _Float16* __restrict__ H, const float* __restrict__ W_out,
                           const float* __restrict__ b_out, float* __restrict__ out8) {
    int idx = blockIdx.x * blockDim.x + threadIdx.x;     // 2048*8 threads
    int o = idx & 7;
    int b = idx >> 3;
    float s = b_out[o];
    const _Float16* h = H + (size_t)b * 512;
    for (int k = 0; k < 512; ++k) s += (float)h[k] * W_out[k * 8 + o];
    out8[idx] = s;
}

// ---- DDIM combine (last step only), joint clamp, DH forward kinematics, error
__global__ void k_fk(const float* __restrict__ out8, const float* __restrict__ noise_s,
                     const float* __restrict__ x_pre, float zc_nz, float zc_j,
                     float* __restrict__ joints_s, float* __restrict__ err_s) {
    int b = blockIdx.x * blockDim.x + threadIdx.x;
    if (b >= 2048) return;
    const float PI = 3.14159265358979323846f;
    const float high[8] = { 35.f*PI/180.f,  -60.f*PI/180.f, 3.94f, 155.f*PI/180.f,
                            -55.f*PI/180.f,  PI,             5.f*PI/180.f, 3.71f };
    const float low[8]  = {-35.f*PI/180.f, -155.f*PI/180.f, 2.59f,  60.f*PI/180.f,
                           -125.f*PI/180.f, -PI,           -90.f*PI/180.f, 2.5f };
    float joint[8];
#pragma unroll
    for (int i = 0; i < 8; ++i) {
        float nz = noise_s[b * 8 + i];
        float z  = zc_nz * nz + zc_j * out8[b * 8 + i];
        float jn = (tanhf(z * 0.1f) + 1.0f) * 0.5f;
        float v  = jn * (high[i] - low[i]) + low[i];
        joint[i] = v;
        joints_s[b * 8 + i] = v;
    }
    const float alpha_deg[8] = {0.f, -90.f, -90.f, 90.f, -90.f, -90.f, 90.f, -90.f};
    const float aarr[8] = {0.f, 0.16f, 0.07f, 0.f, 0.1334f, 0.f, 0.15f, 0.3625f};
    float darr[8]  = {0.f, 0.f, joint[2], 0.f, -0.1316f, 1.0105f, 0.52f, joint[7]};
    float tharr[8] = {joint[0], joint[1], 0.f, joint[3], joint[4], joint[5], joint[6], 0.f};
    float Tm[16];
    for (int j = 0; j < 8; ++j) {
        float al = alpha_deg[j] * (PI / 180.f);
        float ca = cosf(al), sa = sinf(al);
        float ct = cosf(tharr[j]), st = sinf(tharr[j]);
        float dd = darr[j], aa = aarr[j];
        float M[16];
        M[0] = ct;      M[1] = -st;     M[2] = 0.f;  M[3] = aa;
        M[4] = st * ca; M[5] = ct * ca; M[6] = -sa;  M[7] = -sa * dd;
        M[8] = st * sa; M[9] = ct * sa; M[10] = ca;  M[11] = ca * dd;
        M[12] = 0.f;    M[13] = 0.f;    M[14] = 0.f; M[15] = 1.f;
        if (j == 0) {
            for (int e = 0; e < 16; ++e) Tm[e] = M[e];
        } else {
            float Tn[16];
            for (int r = 0; r < 4; ++r)
                for (int c = 0; c < 4; ++c) {
                    float s = 0.f;
                    for (int q = 0; q < 4; ++q) s += Tm[r * 4 + q] * M[q * 4 + c];
                    Tn[r * 4 + c] = s;
                }
            for (int e = 0; e < 16; ++e) Tm[e] = Tn[e];
        }
    }
    float e1x = 3.75f * Tm[2], e1y = 3.75f * Tm[6], e1z = 3.75f * Tm[10];
    float p[6] = { Tm[3], Tm[7], Tm[11] + 1.702f,
                   e1x + Tm[3], e1y + Tm[7], e1z + Tm[11] + 1.702f };
    float d0 = 0.f, d1 = 0.f;
    for (int i = 0; i < 3; ++i) { float df = 100.f * p[i]     - x_pre[b * 6 + i];     d0 += df * df; }
    for (int i = 0; i < 3; ++i) { float df = 100.f * p[3 + i] - x_pre[b * 6 + 3 + i]; d1 += df * df; }
    err_s[b] = sqrtf(d0) + sqrtf(d1);
}

// ---- argmin over the 2 steps, gather best joints
__global__ void k_select(const float* __restrict__ joints, const float* __restrict__ err,
                         float* __restrict__ out) {
    int b = blockIdx.x * blockDim.x + threadIdx.x;
    if (b >= 2048) return;
    int src = (err[2048 + b] < err[b]) ? 1 : 0;          // argmin: first index wins ties
#pragma unroll
    for (int i = 0; i < 8; ++i)
        out[b * 8 + i] = joints[(size_t)src * 2048 * 8 + b * 8 + i];
}

extern "C" void kernel_launch(void* const* d_in, const int* in_sizes, int n_in,
                              void* d_out, int out_size, void* d_ws, size_t ws_size,
                              hipStream_t stream) {
    const float* x_pre = (const float*)d_in[0];
    const float* noise = (const float*)d_in[1];
    const float* W_in  = (const float*)d_in[2];
    const float* b_in  = (const float*)d_in[3];
    const float* W1 = (const float*)d_in[4];  const float* b1 = (const float*)d_in[5];
    const float* W2 = (const float*)d_in[6];  const float* b2 = (const float*)d_in[7];
    const float* W3 = (const float*)d_in[8];  const float* b3 = (const float*)d_in[9];
    const float* W4 = (const float*)d_in[10]; const float* b4 = (const float*)d_in[11];
    const float* W_out = (const float*)d_in[12];
    const float* b_out = (const float*)d_in[13];
    float* out = (float*)d_out;

    char* ws = (char*)d_ws;
    _Float16* Wt       = (_Float16*)(ws);                       // 4*512*512*2 = 2 MB
    float*    const_in = (float*)(ws + 2097152);                // 2 KB
    _Float16* hA       = (_Float16*)(ws + 2099200);             // 2 MB
    _Float16* hB       = (_Float16*)(ws + 4196352);             // 2 MB
    float*    out8     = (float*)(ws + 6293504);                // 64 KB
    float*    joints   = (float*)(ws + 6359040);                // 128 KB
    float*    err      = (float*)(ws + 6490112);                // 16 KB

    // DDIM schedule: only the LAST coefficient tuple survives the reference loop.
    // ts = linspace(1000,0,21) -> cur = 49, prev = -1 -> ab_p = 1.
    double ab = 1.0;
    for (int t = 1; t <= 50; ++t)
        ab *= std::exp(-0.1 / 1000.0 - 0.5 * (10.0 - 0.1) * (2.0 * t - 1.0) / 1.0e6);
    double c1 = std::sqrt(1.0 / ab);
    double c2 = -std::sqrt((1.0 - ab) / ab);    // max(1-ab_p-var,0)=0 term vanishes
    double c3 = std::sqrt(1.0 - ab);
    float zc_nz = (float)(c1 + c3);
    float zc_j  = (float)c2;
    float cur_t = 49.0f;

    k_transpose_w<<<4096, 256, 0, stream>>>(W1, W2, W3, W4, Wt);
    k_const_in<<<1, 512, 0, stream>>>(W_in, b_in, const_in, cur_t);

    dim3 ggrid(16, 8);  // 2048/128 x 512/64
    for (int s = 0; s < 2; ++s) {
        const float* nz = noise + (size_t)s * 2048 * 8;
        k_first_layer<<<4096, 256, 0, stream>>>(x_pre, nz, W_in, const_in, hA);
        k_gemm512<<<ggrid, 128, 0, stream>>>(hA, Wt + 0 * 262144, b1, hB);
        k_gemm512<<<ggrid, 128, 0, stream>>>(hB, Wt + 1 * 262144, b2, hA);
        k_gemm512<<<ggrid, 128, 0, stream>>>(hA, Wt + 2 * 262144, b3, hB);
        k_gemm512<<<ggrid, 128, 0, stream>>>(hB, Wt + 3 * 262144, b4, hA);
        k_out_proj<<<64, 256, 0, stream>>>(hA, W_out, b_out, out8);
        k_fk<<<8, 256, 0, stream>>>(out8, nz, x_pre, zc_nz, zc_j,
                                    joints + (size_t)s * 2048 * 8, err + (size_t)s * 2048);
    }
    k_select<<<8, 256, 0, stream>>>(joints, err, out);
}
